// StackNetwork_7181185319185
// MI455X (gfx1250) — matrix operand order repaired and checked by reading, not compiled, not run on hardware
//
#include <hip/hip_runtime.h>
#include <hip/hip_bf16.h>

#define B_   4096
#define T_   64
#define OBS_ 64
#define H_   64
#define A_   8
#define S_   201

typedef _Float16 v16h __attribute__((ext_vector_type(16)));
typedef _Float16 v8h  __attribute__((ext_vector_type(8)));
typedef float    v8f  __attribute__((ext_vector_type(8)));
typedef float    v4f  __attribute__((ext_vector_type(4)));

__device__ __forceinline__ v8f wmma16(v16h a, v16h b, v8f c) {
  // D = A(16x32 f16) * B(32x16 f16) + C(16x16 f32)
  return __builtin_amdgcn_wmma_f32_16x16x32_f16(false, a, false, b,
                                                (short)0, c, false, false);
}

// gfx1250 hardware tanh (TRANS32 op; co-executes with WMMA) instead of the
// branchy ocml exp-based tanhf expansion.
__device__ __forceinline__ float fast_tanh(float v) {
#if __has_builtin(__builtin_amdgcn_tanhf)
  return __builtin_amdgcn_tanhf(v);
#else
  float r;
  asm volatile("v_tanh_f32 %0, %1" : "=v"(r) : "v"(v));
  return r;
#endif
}

// Build A-fragment (16 halves/lane) from a 64-float row in global memory.
// Contiguous k order (not ISA table order): lane holds row m=lane%16; elems 0..15 -> K = k0..k0+15
// (k0 already includes the 16*hi lane-half shift).
__device__ __forceinline__ v16h loadA_f32(const float* rowbase, int k0) {
  const v4f* p = (const v4f*)(rowbase + k0);
  v4f a0 = p[0], a1 = p[1];   // k0 .. k0+7
  v4f a2 = p[2], a3 = p[3];   // k0+8 .. k0+15
  v16h a;
#pragma unroll
  for (int i = 0; i < 4; ++i) {
    a[i]      = (_Float16)a0[i];
    a[4 + i]  = (_Float16)a1[i];
    a[8 + i]  = (_Float16)a2[i];
    a[12 + i] = (_Float16)a3[i];
  }
  return a;
}

// Same, from a 64-half f16 row in LDS (two ds_load_b128).
__device__ __forceinline__ v16h loadA_h(const _Float16* rowbase, int k0) {
  v8h lo = *(const v8h*)(rowbase + k0);
  v8h hh = *(const v8h*)(rowbase + k0 + 8);
  return __builtin_shufflevector(lo, hh, 0,1,2,3,4,5,6,7,8,9,10,11,12,13,14,15);
}

__global__ void stack_copy_kernel(const float4* __restrict__ src,
                                  float4* __restrict__ dst, size_t n4) {
  size_t i = (size_t)blockIdx.x * blockDim.x + threadIdx.x;
  size_t stride = (size_t)gridDim.x * blockDim.x;
  for (; i < n4; i += stride) dst[i] = src[i];
}

__global__ void __launch_bounds__(32)
stacknet_kernel(const float* __restrict__ x,
                const int*   __restrict__ stack_ptrs,
                const float* __restrict__ W1, const float* __restrict__ b1,
                const float* __restrict__ W2, const float* __restrict__ b2,
                const float* __restrict__ Ws, const float* __restrict__ bs,
                const float* __restrict__ Wp, const float* __restrict__ bp,
                const float* __restrict__ Wv, const float* __restrict__ bv,
                float* __restrict__ out_logits,
                float* __restrict__ out_values,
                float*              out_stack,
                int*   __restrict__ out_ptrs)
{
  // f16 weights, row-major: B-fragment for column n, K-chunk kc is the 16
  // contiguous halves at sW[n][kc] (same contiguous k order as the A fragments).
  __shared__ __align__(32) _Float16 sW1[H_][OBS_ + H_];   // 16 KB
  __shared__ __align__(32) _Float16 sW2[H_][H_];          //  8 KB
  __shared__ __align__(32) _Float16 sWh[16][H_];          //  2 KB (rows: 0-2 Ws, 3-10 Wp, 11 Wv, 12-15 zero)
  __shared__ __align__(32) _Float16 sAct[16][H_];         //  2 KB staging (h, then p)
  __shared__ float sB1[H_], sB2[H_], sBh[16];
  __shared__ float sScore[16][4];

  const int lane = threadIdx.x;          // single wave32 per block
  // ---- one-time weight staging (f32 -> f16) ----
  for (int i = lane; i < H_ * (OBS_ + H_); i += 32) sW1[0][i] = (_Float16)W1[i];
  for (int i = lane; i < H_ * H_;          i += 32) sW2[0][i] = (_Float16)W2[i];
  for (int i = lane; i < 16 * H_;          i += 32) {
    int r = i >> 6, k = i & 63;
    float v = (r < 3) ? Ws[r * H_ + k]
            : (r < 11) ? Wp[(r - 3) * H_ + k]
            : (r == 11) ? Wv[k] : 0.f;
    sWh[0][i] = (_Float16)v;
  }
  for (int i = lane; i < H_; i += 32) { sB1[i] = b1[i]; sB2[i] = b2[i]; }
  if (lane < 16)
    sBh[lane] = (lane < 3) ? bs[lane] : (lane < 11) ? bp[lane - 3]
              : (lane == 11) ? bv[0] : 0.f;
  __syncthreads();

  const int n16 = lane & 15;             // column index inside a 16-wide tile
  const int hi  = lane >> 4;             // lane half (C rows 0-7 vs 8-15)
  const int m0  = blockIdx.x * 16;       // first batch row of this wave
  const int bRow = m0 + n16;             // batch row this lane's A-fragments cover

  float b1v[4], b2v[4];
#pragma unroll
  for (int nt = 0; nt < 4; ++nt) {
    b1v[nt] = sB1[16 * nt + n16];
    b2v[nt] = sB2[16 * nt + n16];
  }
  const float hbv = sBh[n16];

  int myptr = (lane < 16) ? stack_ptrs[m0 + lane] : 0;

  const float* xrow = x + (size_t)bRow * T_ * OBS_;   // advances by OBS_ per step

  for (int t = 0; t < T_; ++t) {
    // ---- gather A fragments: [x_t | top] (K = 0..127) ----
    int ptrA = __shfl(myptr, n16);       // ptr of batch row n16 (lanes 0-15 own ptrs)
    const float* srow = out_stack + ((size_t)bRow * S_ + ptrA) * H_;
    v16h afrag[4];
    afrag[0] = loadA_f32(xrow, 16 * hi);
    afrag[1] = loadA_f32(xrow, 32 + 16 * hi);
    afrag[2] = loadA_f32(srow, 16 * hi);
    afrag[3] = loadA_f32(srow, 32 + 16 * hi);
    // next step's x row is ptr-independent: pull it off the serial chain
    if (t + 1 < T_) __builtin_prefetch(xrow + OBS_, 0, 0);

    // ---- h = tanh([x,top] @ W1^T + b1): 4 N-tiles x 4 K-tiles ----
#pragma unroll
    for (int nt = 0; nt < 4; ++nt) {
      v8f c = {};
#pragma unroll
      for (int kt = 0; kt < 4; ++kt) {
        v16h bf = *(const v16h*)&sW1[16 * nt + n16][32 * kt + 16 * hi];
        c = wmma16(afrag[kt], bf, c);
      }
#pragma unroll
      for (int r = 0; r < 8; ++r) {
        float hv = fast_tanh(c[r] + b1v[nt]);
        sAct[r + 8 * hi][16 * nt + n16] = (_Float16)hv;   // D-layout -> row-major
      }
    }
    __syncthreads();   // single-wave WG: cheap fence ordering LDS store->load

    // ---- p = tanh(h @ W2^T + b2): 4 N-tiles x 2 K-tiles ----
    v16h ah0 = loadA_h(&sAct[n16][0], 16 * hi);
    v16h ah1 = loadA_h(&sAct[n16][0], 32 + 16 * hi);
    v8f cp[4];
#pragma unroll
    for (int nt = 0; nt < 4; ++nt) {
      v8f c = {};
      c = wmma16(ah0, *(const v16h*)&sW2[16 * nt + n16][16 * hi], c);
      c = wmma16(ah1, *(const v16h*)&sW2[16 * nt + n16][32 + 16 * hi], c);
#pragma unroll
      for (int r = 0; r < 8; ++r) c[r] = fast_tanh(c[r] + b2v[nt]);
      cp[nt] = c;
    }
    // stage p as f16 for the head matmul (reuse sAct)
#pragma unroll
    for (int nt = 0; nt < 4; ++nt)
#pragma unroll
      for (int r = 0; r < 8; ++r)
        sAct[r + 8 * hi][16 * nt + n16] = (_Float16)cp[nt][r];
    __syncthreads();

    // ---- fused head: [stack(3) | policy(8) | value(1)] = p @ Wh^T + bh ----
    v16h ap0 = loadA_h(&sAct[n16][0], 16 * hi);
    v16h ap1 = loadA_h(&sAct[n16][0], 32 + 16 * hi);
    v8f cs = {};
    cs = wmma16(ap0, *(const v16h*)&sWh[n16][16 * hi], cs);
    cs = wmma16(ap1, *(const v16h*)&sWh[n16][32 + 16 * hi], cs);
#pragma unroll
    for (int r = 0; r < 8; ++r) cs[r] += hbv;

    if (n16 < 3) {                        // stack-op scores -> LDS for argmax
#pragma unroll
      for (int r = 0; r < 8; ++r) sScore[r + 8 * hi][n16] = cs[r];
    }
    if (n16 >= 3 && n16 < 11) {           // logits[b,t,a]
#pragma unroll
      for (int r = 0; r < 8; ++r) {
        int m = r + 8 * hi;
        out_logits[((size_t)(m0 + m) * T_ + t) * A_ + (n16 - 3)] = cs[r];
      }
    } else if (n16 == 11) {               // values[t*B + b] (time-major)
#pragma unroll
      for (int r = 0; r < 8; ++r) {
        int m = r + 8 * hi;
        out_values[(size_t)t * B_ + (m0 + m)] = cs[r];
      }
    }
    __syncthreads();

    // ---- op = argmax(scores) (softmax is monotone), pointer update ----
    int oldp = myptr;
    int push = 0;
    if (lane < 16) {
      float s0 = sScore[lane][0], s1 = sScore[lane][1], s2 = sScore[lane][2];
      int op = 0; float best = s0;
      if (s1 > best) { best = s1; op = 1; }
      if (s2 > best) { op = 2; }
      push = (op == 2) ? 1 : 0;
      int np = myptr + op - 1;
      myptr = np < 0 ? 0 : np;
    }
    unsigned long long pmask = __ballot(push);

    // ---- conditional scatter: push rows write p at OLD ptr ----
    // (writing `top` back for non-push rows is an identity -> skipped)
#pragma unroll
    for (int r = 0; r < 8; ++r) {
      int m = r + 8 * hi;
      int pm = __shfl(oldp, m);           // converged shuffle, src lanes 0-15
      if ((pmask >> m) & 1ull) {
        float* drow = out_stack + ((size_t)(m0 + m) * S_ + pm) * H_;
#pragma unroll
        for (int nt = 0; nt < 4; ++nt) drow[16 * nt + n16] = cp[nt][r];
      }
    }

    xrow += OBS_;
  }

  if (lane < 16) out_ptrs[m0 + lane] = myptr;
}

extern "C" void kernel_launch(void* const* d_in, const int* in_sizes, int n_in,
                              void* d_out, int out_size, void* d_ws, size_t ws_size,
                              hipStream_t stream) {
  const float* x      = (const float*)d_in[0];
  const float* stack0 = (const float*)d_in[1];
  const int*   sptrs  = (const int*)d_in[2];
  const float* W1 = (const float*)d_in[3];
  const float* b1 = (const float*)d_in[4];
  const float* W2 = (const float*)d_in[5];
  const float* b2 = (const float*)d_in[6];
  const float* Ws = (const float*)d_in[7];
  const float* bs = (const float*)d_in[8];
  const float* Wp = (const float*)d_in[9];
  const float* bp = (const float*)d_in[10];
  const float* Wv = (const float*)d_in[11];
  const float* bv = (const float*)d_in[12];

  float* out        = (float*)d_out;
  float* out_logits = out;                               // B*T*A
  float* out_values = out_logits + (size_t)B_ * T_ * A_; // T*B
  float* out_stack  = out_values + (size_t)T_ * B_;      // B*S*H
  int*   out_ptrs   = (int*)(out_stack + (size_t)B_ * S_ * H_); // B

  // 1) copy input stack into the output stack (recurrence mutates it in place)
  size_t n4 = (size_t)B_ * S_ * H_ / 4;
  stack_copy_kernel<<<2048, 256, 0, stream>>>(
      (const float4*)stack0, (float4*)out_stack, n4);

  // 2) recurrence: one wave32 per 16 batch rows -> 256 single-wave workgroups
  stacknet_kernel<<<B_ / 16, 32, 0, stream>>>(
      x, sptrs, W1, b1, W2, b2, Ws, bs, Wp, bp, Wv, bv,
      out_logits, out_values, out_stack, out_ptrs);
}
